// EncoderLayer_62723702390949
// MI455X (gfx1250) — matrix operand order repaired and checked by reading, not compiled, not run on hardware
//
#include <hip/hip_runtime.h>
#include <cstdint>

// ---------------------------------------------------------------------------
// Transformer encoder layer for MI455X (gfx1250), wave32 + WMMA.
// GEMMs: v_wmma_f32_16x16x32_f16, 256x128 block tile, 64x64 wave tile,
// double-buffered LDS filled with global_load_async_to_lds_b128 (ASYNCcnt).
// Weights pre-transposed to [N][K]; V stored transposed per head so flash
// attention reads every LDS fragment with contiguous ds_load_b128.
// ---------------------------------------------------------------------------

typedef _Float16 half_t;
typedef __attribute__((ext_vector_type(16))) _Float16 v16h;
typedef __attribute__((ext_vector_type(8)))  _Float16 v8h;
typedef __attribute__((ext_vector_type(4)))  _Float16 v4h;
typedef __attribute__((ext_vector_type(8)))  float    v8f;

#define D_MODEL 1024
#define HEADS   16
#define DK      64
#define D_FF    4096
#define SEQ     2048
#define BATCH   4
#define ROWS    (BATCH * SEQ)   // 8192

// --------------------- CDNA5 async copy (ASYNCcnt path) --------------------
__device__ __forceinline__ void async_copy16(half_t* lds, const half_t* g) {
  uint32_t loff = (uint32_t)(uintptr_t)lds;   // LDS offset = ADDR[31:0]
  asm volatile("global_load_async_to_lds_b128 %0, %1, off"
               :: "v"(loff), "v"((uint64_t)(uintptr_t)g)
               : "memory");
}
__device__ __forceinline__ void wait_async0() {
  asm volatile("s_wait_asynccnt 0" ::: "memory");
}

// ------------------------- WMMA fragment helpers ---------------------------
// A fragment: 16x32 f16, src row-major [row][k], lda in halves.
__device__ __forceinline__ v16h load_A_frag(const half_t* src, int lda) {
  const int lane = threadIdx.x & 31;
  const int row  = lane & 15;
  const int kb   = (lane >> 4) << 4;           // 0 or 16
  const half_t* p = src + row * lda + kb;
  v16h a;
#pragma unroll
  for (int h = 0; h < 16; ++h) a[h] = p[h];
  return a;
}

// B fragment: 32x16 f16 from transposed storage [n][k] (K contiguous).
__device__ __forceinline__ v16h load_B_fragT(const half_t* srcT, int ldb) {
  const int lane = threadIdx.x & 31;
  const int col  = lane & 15;
  const int kb   = (lane >> 4) << 4;           // 0 or 16
  const half_t* p = srcT + col * ldb + kb;
  v16h b;
#pragma unroll
  for (int h = 0; h < 16; ++h) b[h] = p[h];
  return b;
}

__device__ __forceinline__ v8f wmma_f16(v16h a, v16h b, v8f c) {
  return __builtin_amdgcn_wmma_f32_16x16x32_f16(
      false, a, false, b, (short)0, c, false, false);
}

// ---------------------------- f32 -> f16 convert ---------------------------
__global__ void f32_to_f16_kernel(const float* __restrict__ in,
                                  half_t* __restrict__ out, int n4) {
  int i = blockIdx.x * blockDim.x + threadIdx.x;
  if (i >= n4) return;
  float4 v = ((const float4*)in)[i];
  v4h o;
  o[0] = (half_t)v.x; o[1] = (half_t)v.y; o[2] = (half_t)v.z; o[3] = (half_t)v.w;
  ((v4h*)out)[i] = o;
}

// convert + transpose: W[K][N] f32 -> WT[N][K] f16 (one-time, weights only)
__launch_bounds__(256)
__global__ void f32_to_f16_T_kernel(const float* __restrict__ in,
                                    half_t* __restrict__ outT, int K, int N) {
  __shared__ float tile[32][33];
  const int k0 = blockIdx.y * 32, n0 = blockIdx.x * 32;
  const int tx = threadIdx.x & 31, ty = threadIdx.x >> 5;   // 32 x 8
#pragma unroll
  for (int i = 0; i < 32; i += 8)
    tile[ty + i][tx] = in[(size_t)(k0 + ty + i) * N + n0 + tx];
  __syncthreads();
#pragma unroll
  for (int i = 0; i < 32; i += 8)
    outT[(size_t)(n0 + ty + i) * K + k0 + tx] = (half_t)tile[tx][ty + i];
}

// ------------------------------- WMMA GEMM ---------------------------------
// C[M,N] = A[M,K] * BT[N,K]^T + bias. Block 256x128, BK=32, 256 thr = 8 waves
// (4x2), wave tile 64x64 (16 WMMA / K-step). Async double-buffered tiles.
// flags: bit0=ReLU, bit1=head-split [B,H,S,DK], bit2=head-split-T [B,H,DK,S].
#define BM  256
#define BN  128
#define BK  32
#define LDT 40   // LDS leading dim (halves): 16B-aligned rows, bank skew

__launch_bounds__(256)
__global__ void gemm_f16_kernel(const half_t* __restrict__ A,
                                const half_t* __restrict__ BT,
                                const float*  __restrict__ bias,
                                float*  __restrict__ outF,
                                half_t* __restrict__ outH,
                                int M, int N, int K, int flags) {
  __shared__ half_t lsA[2][BM * LDT];   // 2 x 20 KB
  __shared__ half_t lsB[2][BN * LDT];   // 2 x 10 KB  ([n][k] tiles)

  const int tid  = threadIdx.x;
  const int lane = tid & 31;
  const int w    = tid >> 5;     // 0..7
  const int wm   = w >> 1;       // 0..3  (64-row strip)
  const int wn   = w & 1;        // 0..1  (64-col strip)
  const int blockRow = blockIdx.y * BM;
  const int blockCol = blockIdx.x * BN;

  // async chunk mapping: rows of 64B = 4 chunks of 16B
  const int crow = tid >> 2;          // 0..63
  const int coff = (tid & 3) << 3;    // half offset 0/8/16/24

  v8f acc[4][4];
#pragma unroll
  for (int i = 0; i < 4; ++i)
#pragma unroll
    for (int j = 0; j < 4; ++j) acc[i][j] = (v8f)0.0f;

  auto issue = [&](int kt, int buf) {
    const half_t* ga = A  + (size_t)(blockRow + crow) * K + kt + coff;
    const half_t* gb = BT + (size_t)(blockCol + crow) * K + kt + coff;
#pragma unroll
    for (int i = 0; i < 4; ++i)   // 256 A rows
      async_copy16(&lsA[buf][(64 * i + crow) * LDT + coff],
                   ga + (size_t)(64 * i) * K);
#pragma unroll
    for (int i = 0; i < 2; ++i)   // 128 B rows
      async_copy16(&lsB[buf][(64 * i + crow) * LDT + coff],
                   gb + (size_t)(64 * i) * K);
  };

  issue(0, 0);
  const int nkt = K / BK;
  for (int it = 0; it < nkt; ++it) {
    const int cur = it & 1;
    wait_async0();
    __syncthreads();
    if (it + 1 < nkt) issue((it + 1) * BK, 1 - cur);

    v16h bf[4];
#pragma unroll
    for (int ni = 0; ni < 4; ++ni)
      bf[ni] = load_B_fragT(&lsB[cur][(wn * 64 + ni * 16) * LDT], LDT);
#pragma unroll
    for (int mi = 0; mi < 4; ++mi) {
      v16h af = load_A_frag(&lsA[cur][(wm * 64 + mi * 16) * LDT], LDT);
#pragma unroll
      for (int ni = 0; ni < 4; ++ni)
        acc[mi][ni] = wmma_f16(af, bf[ni], acc[mi][ni]);
    }
  }

  // Epilogue: C layout = lanes 0-15 rows r, lanes 16-31 rows r+8 (VGPR r).
  const int colInTile = lane & 15;
  const int rowHalf   = (lane >> 4) << 3;
  const bool relu       = (flags & 1) != 0;
  const bool headsplit  = (flags & 2) != 0;
  const bool headsplitT = (flags & 4) != 0;
#pragma unroll
  for (int mi = 0; mi < 4; ++mi) {
#pragma unroll
    for (int ni = 0; ni < 4; ++ni) {
      const int gcol = blockCol + wn * 64 + ni * 16 + colInTile;
      const float bv = bias ? bias[gcol] : 0.0f;
#pragma unroll
      for (int r = 0; r < 8; ++r) {
        const int grow = blockRow + wm * 64 + mi * 16 + rowHalf + r;
        float v = acc[mi][ni][r] + bv;
        if (relu) v = fmaxf(v, 0.0f);
        if (outF) outF[(size_t)grow * N + gcol] = v;
        if (outH) {
          size_t oi;
          const int b = grow >> 11, s = grow & (SEQ - 1);
          const int h = gcol >> 6,  d = gcol & (DK - 1);
          if (headsplit) {          // [B,H,S,DK]
            oi = ((size_t)(b * HEADS + h) * SEQ + s) * DK + d;
          } else if (headsplitT) {  // [B,H,DK,S]  (V^T for attention)
            oi = ((size_t)(b * HEADS + h) * DK + d) * SEQ + s;
          } else {
            oi = (size_t)grow * N + gcol;
          }
          outH[oi] = (half_t)v;
        }
      }
    }
  }
}

// --------------------------- Flash attention -------------------------------
// grid = (SEQ/64, BATCH*HEADS), block = 128 (4 waves). Each wave owns 16
// query rows; streams 64-key K / V^T blocks (async double buffered) with
// online softmax. All LDS fragment reads are contiguous b128.
#define ALD 72

__launch_bounds__(128)
__global__ void attn_kernel(const half_t* __restrict__ Qh,
                            const half_t* __restrict__ Kh,
                            const half_t* __restrict__ Vt,   // [B,H,DK,S]
                            half_t* __restrict__ concatH) {
  __shared__ half_t lsQ[64 * ALD];
  __shared__ half_t lsK[2][64 * ALD];      // [key][d]
  __shared__ half_t lsV[2][64 * ALD];      // [d][key]  (V^T tile)
  __shared__ half_t lsP[4 * 16 * ALD];     // per-wave P staging (C->A relayout)

  const int tid  = threadIdx.x;
  const int lane = tid & 31;
  const int w    = tid >> 5;        // 0..3
  const int bh   = blockIdx.y;      // b*HEADS + h
  const int q0   = blockIdx.x * 64;

  const size_t headBase = (size_t)bh * SEQ * DK;
  const int crow = tid >> 2;          // 0..31
  const int coff = (tid & 3) << 3;    // 0/8/16/24 halves

  auto issue_kv = [&](int kb, int buf) {
    const half_t* gk = Kh + headBase + (size_t)(kb + crow) * DK + coff;
    const half_t* gv = Vt + headBase + (size_t)crow * SEQ + kb + coff;
    async_copy16(&lsK[buf][crow * ALD + coff], gk);
    async_copy16(&lsV[buf][crow * ALD + coff], gv);
    async_copy16(&lsK[buf][(32 + crow) * ALD + coff], gk + (size_t)32 * DK);
    async_copy16(&lsV[buf][(32 + crow) * ALD + coff], gv + (size_t)32 * SEQ);
  };

  {   // Q block 64x64 (async, one-time)
    const half_t* gq = Qh + headBase + (size_t)(q0 + crow) * DK + coff;
    async_copy16(&lsQ[crow * ALD + coff], gq);
    async_copy16(&lsQ[(32 + crow) * ALD + coff], gq + (size_t)32 * DK);
  }
  issue_kv(0, 0);
  wait_async0();
  __syncthreads();

  v16h aQ[2];
  aQ[0] = load_A_frag(&lsQ[(w * 16) * ALD], ALD);
  aQ[1] = load_A_frag(&lsQ[(w * 16) * ALD + 32], ALD);

  float mrun[8], lrun[8];
  v8f oacc[4];
#pragma unroll
  for (int r = 0; r < 8; ++r) { mrun[r] = -1e30f; lrun[r] = 0.0f; }
#pragma unroll
  for (int d = 0; d < 4; ++d) oacc[d] = (v8f)0.0f;

  const int colInTile = lane & 15;
  const int rowHalf   = (lane >> 4) << 3;
  const int NKB = SEQ / 64;   // 32

  for (int it = 0; it < NKB; ++it) {
    const int cur = it & 1;
    if (it > 0) { wait_async0(); __syncthreads(); }
    if (it + 1 < NKB) issue_kv((it + 1) * 64, 1 - cur);

    // scores S = Q K^T : K stored [key][d], fragT reads it directly
    v8f sacc[4];
#pragma unroll
    for (int ni = 0; ni < 4; ++ni) {
      sacc[ni] = (v8f)0.0f;
#pragma unroll
      for (int ks = 0; ks < 2; ++ks) {
        v16h bK = load_B_fragT(&lsK[cur][(ni * 16) * ALD + ks * 32], ALD);
        sacc[ni] = wmma_f16(aQ[ks], bK, sacc[ni]);
      }
    }

    // online softmax: row reductions across 16-lane halves
    float p[4][8];
#pragma unroll
    for (int r = 0; r < 8; ++r) {
      float mx = -1e30f;
#pragma unroll
      for (int ni = 0; ni < 4; ++ni) {
        sacc[ni][r] *= 0.125f;                    // 1/sqrt(DK)
        mx = fmaxf(mx, sacc[ni][r]);
      }
#pragma unroll
      for (int off = 8; off >= 1; off >>= 1)
        mx = fmaxf(mx, __shfl_xor(mx, off, 32));
      const float mnew  = fmaxf(mrun[r], mx);
      const float alpha = __expf(mrun[r] - mnew);
      float rs = 0.0f;
#pragma unroll
      for (int ni = 0; ni < 4; ++ni) {
        p[ni][r] = __expf(sacc[ni][r] - mnew);
        rs += p[ni][r];
      }
#pragma unroll
      for (int off = 8; off >= 1; off >>= 1)
        rs += __shfl_xor(rs, off, 32);
      lrun[r] = lrun[r] * alpha + rs;
      mrun[r] = mnew;
#pragma unroll
      for (int d = 0; d < 4; ++d) oacc[d][r] *= alpha;
    }

    // stage P (C layout) -> LDS -> A-fragment layout; wave-local, DScnt only
    half_t* myP = &lsP[w * 16 * ALD];
#pragma unroll
    for (int ni = 0; ni < 4; ++ni)
#pragma unroll
      for (int r = 0; r < 8; ++r)
        myP[(rowHalf + r) * ALD + ni * 16 + colInTile] = (half_t)p[ni][r];
    asm volatile("s_wait_dscnt 0" ::: "memory");

    v16h aP0 = load_A_frag(myP, ALD);
    v16h aP1 = load_A_frag(myP + 32, ALD);
    // P x V with V^T tile: B frag cols = d, K = keys (contiguous in LDS)
#pragma unroll
    for (int d = 0; d < 4; ++d) {
      oacc[d] = wmma_f16(aP0, load_B_fragT(&lsV[cur][(d * 16) * ALD], ALD), oacc[d]);
      oacc[d] = wmma_f16(aP1, load_B_fragT(&lsV[cur][(d * 16) * ALD + 32], ALD), oacc[d]);
    }
  }

  // normalize and write concat [B,S,D] (f16, feeds Wo GEMM)
  const int b = bh >> 4;
  const int h = bh & 15;
#pragma unroll
  for (int d = 0; d < 4; ++d) {
#pragma unroll
    for (int r = 0; r < 8; ++r) {
      const int s = q0 + w * 16 + rowHalf + r;
      const float v = oacc[d][r] / lrun[r];
      const size_t oi =
          ((size_t)(b * SEQ + s)) * D_MODEL + h * DK + d * 16 + colInTile;
      concatH[oi] = (half_t)v;
    }
  }
}

// ------------------- residual add + LayerNorm (torch-style) ----------------
__launch_bounds__(256)
__global__ void add_norm_kernel(const float* __restrict__ a,
                                const float* __restrict__ b,
                                const float* __restrict__ alpha,
                                const float* __restrict__ bias,
                                float*  __restrict__ outF,
                                half_t* __restrict__ outH) {
  __shared__ float red[16];
  const int row = blockIdx.x;
  const size_t base = (size_t)row * D_MODEL;
  float v[4];
  float s = 0.0f, ss = 0.0f;
#pragma unroll
  for (int j = 0; j < 4; ++j) {
    const int i = j * 256 + threadIdx.x;
    v[j] = a[base + i] + b[base + i];
    s += v[j]; ss += v[j] * v[j];
  }
#pragma unroll
  for (int off = 16; off >= 1; off >>= 1) {
    s  += __shfl_xor(s, off, 32);
    ss += __shfl_xor(ss, off, 32);
  }
  if ((threadIdx.x & 31) == 0) {
    red[threadIdx.x >> 5] = s;
    red[8 + (threadIdx.x >> 5)] = ss;
  }
  __syncthreads();
  s = 0.0f; ss = 0.0f;
#pragma unroll
  for (int i = 0; i < 8; ++i) { s += red[i]; ss += red[8 + i]; }
  const float mean = s * (1.0f / D_MODEL);
  const float var  = (ss - (float)D_MODEL * mean * mean) * (1.0f / (D_MODEL - 1));
  const float inv  = 1.0f / (sqrtf(fmaxf(var, 0.0f)) + 1e-6f);
#pragma unroll
  for (int j = 0; j < 4; ++j) {
    const int i = j * 256 + threadIdx.x;
    const float o = alpha[i] * (v[j] - mean) * inv + bias[i];
    outF[base + i] = o;
    if (outH) outH[base + i] = (half_t)o;
  }
}

// ------------------------------- launcher ----------------------------------
extern "C" void kernel_launch(void* const* d_in, const int* in_sizes, int n_in,
                              void* d_out, int out_size, void* d_ws, size_t ws_size,
                              hipStream_t stream) {
  (void)in_sizes; (void)n_in; (void)out_size; (void)ws_size;

  const float* x      = (const float*)d_in[0];
  const float* Wq     = (const float*)d_in[1];
  const float* bq     = (const float*)d_in[2];
  const float* Wk     = (const float*)d_in[3];
  const float* bk     = (const float*)d_in[4];
  const float* Wv     = (const float*)d_in[5];
  const float* bv     = (const float*)d_in[6];
  const float* Wo     = (const float*)d_in[7];
  const float* bo     = (const float*)d_in[8];
  const float* alpha1 = (const float*)d_in[9];
  const float* bias1  = (const float*)d_in[10];
  const float* alpha2 = (const float*)d_in[11];
  const float* bias2  = (const float*)d_in[12];
  const float* W1     = (const float*)d_in[13];
  const float* b1     = (const float*)d_in[14];
  const float* W2     = (const float*)d_in[15];
  const float* b2     = (const float*)d_in[16];

  char* ws = (char*)d_ws;
  // lifetime-aliased workspace layout (peak ~168 MB); *T = [N][K] transposed
  half_t* WqT     = (half_t*)(ws + 0);
  half_t* WkT     = (half_t*)(ws + 2097152);
  half_t* WvT     = (half_t*)(ws + 4194304);
  half_t* WoT     = (half_t*)(ws + 6291456);
  half_t* W1T     = (half_t*)(ws + 8388608);
  half_t* W2T     = (half_t*)(ws + 16777216);
  half_t* xh      = (half_t*)(ws + 25165824);   // dead after QKV proj
  half_t* norm1H  = (half_t*)(ws + 25165824);   // reuses xh slot
  half_t* Qh      = (half_t*)(ws + 41943040);
  half_t* Kh      = (half_t*)(ws + 58720256);
  half_t* VtH     = (half_t*)(ws + 75497472);   // V^T per head
  half_t* concatH = (half_t*)(ws + 92274688);
  half_t* ff1H    = (half_t*)(ws + 41943040);   // reuses Q/K/V + concat (67MB)
  float*  attnOut = (float*)(ws + 109051904);
  float*  ff2     = (float*)(ws + 109051904);   // reuses attnOut
  float*  norm1F  = (float*)(ws + 142606336);
  float*  out     = (float*)d_out;

  // --- convert x; convert+transpose weights ---
  const int CB = 256;
  f32_to_f16_kernel<<<(ROWS * D_MODEL / 4 + CB - 1) / CB, CB, 0, stream>>>(
      x, xh, ROWS * D_MODEL / 4);
  f32_to_f16_T_kernel<<<dim3(32, 32),  256, 0, stream>>>(Wq, WqT, D_MODEL, D_MODEL);
  f32_to_f16_T_kernel<<<dim3(32, 32),  256, 0, stream>>>(Wk, WkT, D_MODEL, D_MODEL);
  f32_to_f16_T_kernel<<<dim3(32, 32),  256, 0, stream>>>(Wv, WvT, D_MODEL, D_MODEL);
  f32_to_f16_T_kernel<<<dim3(32, 32),  256, 0, stream>>>(Wo, WoT, D_MODEL, D_MODEL);
  f32_to_f16_T_kernel<<<dim3(128, 32), 256, 0, stream>>>(W1, W1T, D_MODEL, D_FF);
  f32_to_f16_T_kernel<<<dim3(32, 128), 256, 0, stream>>>(W2, W2T, D_FF, D_MODEL);

  // --- Q/K/V projections (Q,K head-split; V head-split transposed) ---
  dim3 gProj(D_MODEL / BN, ROWS / BM);   // (8, 32)
  gemm_f16_kernel<<<gProj, 256, 0, stream>>>(xh, WqT, bq, nullptr, Qh,
                                             ROWS, D_MODEL, D_MODEL, 2);
  gemm_f16_kernel<<<gProj, 256, 0, stream>>>(xh, WkT, bk, nullptr, Kh,
                                             ROWS, D_MODEL, D_MODEL, 2);
  gemm_f16_kernel<<<gProj, 256, 0, stream>>>(xh, WvT, bv, nullptr, VtH,
                                             ROWS, D_MODEL, D_MODEL, 4);

  // --- flash attention ---
  dim3 gAttn(SEQ / 64, BATCH * HEADS);
  attn_kernel<<<gAttn, 128, 0, stream>>>(Qh, Kh, VtH, concatH);

  // --- output projection (f32 out) ---
  gemm_f16_kernel<<<gProj, 256, 0, stream>>>(concatH, WoT, bo, attnOut, nullptr,
                                             ROWS, D_MODEL, D_MODEL, 0);

  // --- residual + norm 1 (f32 + f16 outputs) ---
  add_norm_kernel<<<ROWS, 256, 0, stream>>>(attnOut, x, alpha1, bias1,
                                            norm1F, norm1H);

  // --- FFN ---
  dim3 gFF1(D_FF / BN, ROWS / BM);       // (32, 32)
  gemm_f16_kernel<<<gFF1, 256, 0, stream>>>(norm1H, W1T, b1, nullptr, ff1H,
                                            ROWS, D_FF, D_MODEL, 1);
  gemm_f16_kernel<<<gProj, 256, 0, stream>>>(ff1H, W2T, b2, ff2, nullptr,
                                             ROWS, D_MODEL, D_FF, 0);

  // --- residual + norm 2 -> final output ---
  add_norm_kernel<<<ROWS, 256, 0, stream>>>(ff2, norm1F, alpha2, bias2,
                                            out, nullptr);
}
